// LinearAttention_10763188044435
// MI455X (gfx1250) — matrix operand order repaired and checked by reading, not compiled, not run on hardware
//
#include <hip/hip_runtime.h>
#include <hip/hip_bf16.h>
#include <stdint.h>

// Problem constants (from reference)
#define B_     2
#define T_     8192
#define H_     8
#define HID_   1024
#define CHUNK_ 64
#define NCH_   128          // T/CHUNK
#define PHID_  256          // 2*DQK feature dim after hedgehog

typedef __attribute__((ext_vector_type(16))) __bf16 v16bf;
typedef __attribute__((ext_vector_type(8)))  float  v8f;
typedef unsigned short u16t;
typedef unsigned int   u32t;

// Native f32 -> bf16 (compiler lowers fptrunc to HW cvt; avoids manual bit math)
__device__ __forceinline__ u16t f2bf(float f) {
  union { __bf16 b; u16t u; } un;
  un.b = (__bf16)f;
  return un.u;
}
__device__ __forceinline__ u32t pack2bf(float x, float y) {
  return (u32t)f2bf(x) | ((u32t)f2bf(y) << 16);
}

__device__ __forceinline__ v8f v8zero() {
  v8f r;
  #pragma unroll
  for (int i = 0; i < 8; ++i) r[i] = 0.0f;
  return r;
}

__device__ __forceinline__ v8f wmma_bf16(v16bf a, v16bf b, v8f c) {
  // D = A(16x32) * B(32x16) + C, f32 accumulate
  return __builtin_amdgcn_wmma_f32_16x16x32_bf16(false, a, false, b, (short)0, c,
                                                 false, false);
}

// A fragment (16x32 bf16). LDS tile is row-major [rows][stride].
// Lane layout: lanes 0-15 hold K {0..7,16..23}, lanes 16-31 hold K {8..15,24..31}.
__device__ __forceinline__ v16bf frag_a(const u16t* lds, int stride, int m0, int k0,
                                        int lane) {
  int row = m0 + (lane & 15);
  int kk  = k0 + ((lane >> 4) << 3);
  const u16t* p = lds + row * stride + kk;
  union { uint4 u[2]; v16bf v; } un;
  un.u[0] = *(const uint4*)(p);        // K kk..kk+7
  un.u[1] = *(const uint4*)(p + 16);   // K kk+16..kk+23
  return un.v;
}

// B fragment (32x16 bf16) read from *transposed* storage Bt[n][k] row-major.
// Lane layout: lanes 0-15 hold K {0..7,16..23}, lanes 16-31 hold K {8..15,24..31}, column = lane&15.
__device__ __forceinline__ v16bf frag_b(const u16t* lds, int stride, int n0, int k0,
                                        int lane) {
  int row = n0 + (lane & 15);
  int kk  = k0 + ((lane >> 4) << 3);
  const u16t* p = lds + row * stride + kk;
  union { uint4 u[2]; v16bf v; } un;
  un.u[0] = *(const uint4*)(p);
  un.u[1] = *(const uint4*)(p + 16);
  return un.v;
}

// ---------------------------------------------------------------------------
// Y[M,N] = X[M,K] @ W[N,K]^T, X/W bf16, output f32 or bf16.
// Block: 256 threads (8 waves), 128x128 output tile, K staged 32 at a time.
// Staging is a pure b128 copy (no conversion VALU).
// ---------------------------------------------------------------------------
template <bool OUTBF>
__global__ __launch_bounds__(256)
void gemm_wmma(const u16t* __restrict__ X, const u16t* __restrict__ W,
               void* __restrict__ Yv, int M, int N, int K) {
  __shared__ u16t As[128 * 32];
  __shared__ u16t Bs[128 * 32];
  const int tid  = threadIdx.x;
  const int lane = tid & 31;
  const int wave = tid >> 5;
  const int bm = blockIdx.x * 128;
  const int bn = blockIdx.y * 128;
  const int wm = (wave >> 2) * 64;   // wave M offset
  const int wn = (wave & 3) * 32;    // wave N offset

  v8f acc[4][2];
  #pragma unroll
  for (int mi = 0; mi < 4; ++mi)
    #pragma unroll
    for (int ni = 0; ni < 2; ++ni) acc[mi][ni] = v8zero();

  const int ksteps = K >> 5;
  for (int kt = 0; kt < ksteps; ++kt) {
    const int k0 = kt * 32;
    // Stage A (X tile) and B (W tile): 512 uint4 each, 2 per thread.
    #pragma unroll
    for (int it = 0; it < 2; ++it) {
      int idx = tid + it * 256;          // 0..511 uint4 slots
      int row = idx >> 2;                // 0..127
      int c8  = (idx & 3) << 3;          // bf16 col 0/8/16/24
      *(uint4*)&As[row * 32 + c8] = *(const uint4*)(X + (size_t)(bm + row) * K + k0 + c8);
      *(uint4*)&Bs[row * 32 + c8] = *(const uint4*)(W + (size_t)(bn + row) * K + k0 + c8);
    }
    if (kt + 1 < ksteps) {               // global_prefetch_b8 of next K tile
      __builtin_prefetch(X + (size_t)(bm + (tid >> 1)) * K + k0 + 32, 0, 3);
      __builtin_prefetch(W + (size_t)(bn + (tid >> 1)) * K + k0 + 32, 0, 3);
    }
    __syncthreads();

    v16bf bF[2];
    #pragma unroll
    for (int ni = 0; ni < 2; ++ni) bF[ni] = frag_b(Bs, 32, wn + ni * 16, 0, lane);
    #pragma unroll
    for (int mi = 0; mi < 4; ++mi) {
      v16bf aF = frag_a(As, 32, wm + mi * 16, 0, lane);
      acc[mi][0] = wmma_bf16(aF, bF[0], acc[mi][0]);
      acc[mi][1] = wmma_bf16(aF, bF[1], acc[mi][1]);
    }
    __syncthreads();
  }

  // Epilogue: C layout = VGPR r -> M = r + 8*(lane>=16), N = lane&15.
  #pragma unroll
  for (int mi = 0; mi < 4; ++mi)
    #pragma unroll
    for (int ni = 0; ni < 2; ++ni) {
      int m0 = bm + wm + mi * 16 + ((lane >> 4) << 3);
      int n  = bn + wn + ni * 16 + (lane & 15);
      #pragma unroll
      for (int r = 0; r < 8; ++r) {
        if (OUTBF)
          ((u16t*)Yv)[(size_t)(m0 + r) * N + n] = f2bf(acc[mi][ni][r]);
        else
          ((float*)Yv)[(size_t)(m0 + r) * N + n] = acc[mi][ni][r];
      }
    }
}

// ---------------------------------------------------------------------------
// Hedgehog feature map: per (b,t,h) row (M' = B*T*H rows of 128):
//   y = row @ fw^T + fb;  phi = outscale * softmax(concat(2y, -2y))  -> 256 bf16
// inB / fwB already bf16. Block: 256 threads, 64 rows, weight rows 0..63 in LDS.
// ---------------------------------------------------------------------------
__global__ __launch_bounds__(256)
void hedgehog_wmma(const u16t* __restrict__ inB, const u16t* __restrict__ fwB,
                   const float* __restrict__ fb, u16t* __restrict__ phi,
                   float outscale) {
  extern __shared__ char smem[];
  u16t*  Ws  = (u16t*)smem;                              // 128*128 bf16 (32KB)
  u16t*  Qs  = (u16t*)(smem + 32768);                    // 64*128 bf16  (16KB)
  float* Ys  = (float*)(smem + 32768 + 16384);           // 64*128 f32   (32KB)
  float* red = (float*)(smem + 32768 + 16384 + 32768);   // 64*4 partial sums

  const int tid  = threadIdx.x;
  const int lane = tid & 31;
  const int wave = tid >> 5;
  const int row0 = blockIdx.x * 64;

  // Copy feature-map weight rows 0..63 (1024 uint4) and 32 input rows (512 uint4).
  #pragma unroll
  for (int it = 0; it < 4; ++it)
    ((uint4*)Ws)[tid + it * 256] = ((const uint4*)fwB)[tid + it * 256];
  #pragma unroll
  for (int it = 0; it < 2; ++it) {
    int i = tid + it * 256;
    ((uint4*)Qs)[i] = *(const uint4*)(inB + (size_t)row0 * 128 + (size_t)i * 8);
  }
  __syncthreads();

  // y = q @ fw^T : 64x128, wave w -> m-tile (w>>1), n-half (w&1)*64.
  const int mt = wave >> 1;
  const int nb = (wave & 1) * 64;
  v8f acc[4];
  #pragma unroll
  for (int ni = 0; ni < 4; ++ni) acc[ni] = v8zero();
  #pragma unroll
  for (int kt = 0; kt < 4; ++kt) {
    v16bf aF = frag_a(Qs, 128, mt * 16, kt * 32, lane);
    #pragma unroll
    for (int ni = 0; ni < 4; ++ni) {
      v16bf bF = frag_b(Ws, 128, nb + ni * 16, kt * 32, lane);
      acc[ni] = wmma_bf16(aF, bF, acc[ni]);
    }
  }
  #pragma unroll
  for (int ni = 0; ni < 4; ++ni) {
    int n = nb + ni * 16 + (lane & 15);
    float bias = fb[n];
    int m0 = mt * 16 + ((lane >> 4) << 3);
    #pragma unroll
    for (int r = 0; r < 8; ++r)
      Ys[(m0 + r) * 128 + n] = 2.0f * (acc[ni][r] + bias);   // store 2y
  }
  __syncthreads();

  // Softmax over concat(2y, -2y): 4 threads per row, 32 cols each.
  const int row = tid >> 2;
  const int seg = (tid & 3) * 32;
  float s = 0.0f;
  #pragma unroll
  for (int j = 0; j < 32; ++j) {
    float y = Ys[row * 128 + seg + j];
    s += __expf(y) + __expf(-y);
  }
  red[row * 4 + (tid & 3)] = s;
  __syncthreads();
  float Z = red[row * 4 + 0] + red[row * 4 + 1] + red[row * 4 + 2] + red[row * 4 + 3];
  float inv = outscale / Z;
  size_t gr = (size_t)(row0 + row) * PHID_;
  #pragma unroll
  for (int j = 0; j < 32; j += 2) {
    float y0 = Ys[row * 128 + seg + j];
    float y1 = Ys[row * 128 + seg + j + 1];
    *(u32t*)&phi[gr + seg + j] = pack2bf(__expf(y0) * inv, __expf(y1) * inv);
    *(u32t*)&phi[gr + 128 + seg + j] = pack2bf(__expf(-y0) * inv, __expf(-y1) * inv);
  }
}

// ---------------------------------------------------------------------------
// Chunked linear attention scan. One block per (b, h, dv-slice of 32).
// State S[256][32] f32 lives in registers, distributed over 8 waves
// (wave w owns d-tiles {2w,2w+1}, both 16-col v tiles).
// ---------------------------------------------------------------------------
__global__ __launch_bounds__(256)
void chunk_attn_wmma(const u16t* __restrict__ phiQ, const u16t* __restrict__ phiK,
                     const u16t* __restrict__ vB, float* __restrict__ oW) {
  extern __shared__ char smem[];
  u16t* qS = (u16t*)smem;           // [64][256] phi_q chunk        (32KB)
  u16t* kS = qS + 64 * 256;         // [64][256] phi_k chunk        (32KB)
  u16t* kT = kS + 64 * 256;         // [256][64] phi_k transposed   (32KB)
  u16t* vT = kT + 256 * 64;         // [32][64]  v slice transposed ( 4KB)
  u16t* sT = vT + 32 * 64;          // [32][256] S^T as bf16        (16KB)
  u16t* aB = sT + 32 * 256;         // [64][64]  masked scores bf16 ( 8KB)

  const int tid  = threadIdx.x;
  const int lane = tid & 31;
  const int wave = tid >> 5;
  const int bh = blockIdx.x >> 2;
  const int vs = blockIdx.x & 3;
  const int b  = bh >> 3;
  const int h  = bh & 7;

  v8f S[2][2];
  #pragma unroll
  for (int mi = 0; mi < 2; ++mi)
    #pragma unroll
    for (int ni = 0; ni < 2; ++ni) S[mi][ni] = v8zero();

  for (int c = 0; c < NCH_; ++c) {
    const int t0 = c * CHUNK_;

    // ---- stage chunk into LDS ----
    #pragma unroll
    for (int it = 0; it < 8; ++it) {
      int i   = tid + it * 256;            // uint4 idx 0..2047 (32 per row)
      int row = i >> 5;
      int c8  = (i & 31) << 3;
      size_t g = ((size_t)(b * T_ + t0 + row) * H_ + h) * PHID_ + c8;
      ((uint4*)qS)[i] = *(const uint4*)(phiQ + g);
      union { uint4 u; u16t s[8]; } kk;
      kk.u = *(const uint4*)(phiK + g);
      ((uint4*)kS)[i] = kk.u;
      #pragma unroll
      for (int e = 0; e < 8; ++e) kT[(c8 + e) * 64 + row] = kk.s[e];
    }
    {
      int row = tid >> 2;
      int j0  = (tid & 3) << 3;
      size_t g = (size_t)(b * T_ + t0 + row) * HID_ + h * 128 + vs * 32 + j0;
      union { uint4 u; u16t s[8]; } vv;
      vv.u = *(const uint4*)(vB + g);
      #pragma unroll
      for (int e = 0; e < 8; ++e) vT[(j0 + e) * 64 + row] = vv.s[e];
    }
    // S -> sT (bf16, [vcol][d]) for the q@S product.
    #pragma unroll
    for (int mi = 0; mi < 2; ++mi)
      #pragma unroll
      for (int ni = 0; ni < 2; ++ni) {
        int m0 = (2 * wave + mi) * 16 + ((lane >> 4) << 3);
        int n  = ni * 16 + (lane & 15);
        #pragma unroll
        for (int r = 0; r < 8; ++r) sT[n * 256 + m0 + r] = f2bf(S[mi][ni][r]);
      }
    __syncthreads();

    // ---- intra-chunk scores: attn = q @ k^T, causal-masked, -> bf16 LDS ----
    {
      const int mt = wave >> 1;
      const int nb = (wave & 1) * 2;
      v8f at[2];
      at[0] = v8zero(); at[1] = v8zero();
      #pragma unroll
      for (int ks = 0; ks < 8; ++ks) {
        v16bf aF = frag_a(qS, 256, mt * 16, ks * 32, lane);
        #pragma unroll
        for (int j = 0; j < 2; ++j) {
          v16bf bF = frag_b(kS, 256, (nb + j) * 16, ks * 32, lane);
          at[j] = wmma_bf16(aF, bF, at[j]);
        }
      }
      #pragma unroll
      for (int j = 0; j < 2; ++j) {
        int m0 = mt * 16 + ((lane >> 4) << 3);
        int n  = (nb + j) * 16 + (lane & 15);
        #pragma unroll
        for (int r = 0; r < 8; ++r) {
          int m = m0 + r;
          aB[m * 64 + n] = (n <= m) ? f2bf(at[j][r]) : (u16t)0;
        }
      }
    }
    __syncthreads();

    // ---- o = attn @ v + q @ S ----
    {
      const int mt = wave >> 1;
      const int nt = wave & 1;
      v8f oc = v8zero();
      #pragma unroll
      for (int ks = 0; ks < 2; ++ks) {
        v16bf aF = frag_a(aB, 64, mt * 16, ks * 32, lane);
        v16bf bF = frag_b(vT, 64, nt * 16, ks * 32, lane);
        oc = wmma_bf16(aF, bF, oc);
      }
      #pragma unroll
      for (int ks = 0; ks < 8; ++ks) {
        v16bf aF = frag_a(qS, 256, mt * 16, ks * 32, lane);
        v16bf bF = frag_b(sT, 256, nt * 16, ks * 32, lane);
        oc = wmma_bf16(aF, bF, oc);
      }
      int m0  = t0 + mt * 16 + ((lane >> 4) << 3);
      int col = h * 128 + vs * 32 + nt * 16 + (lane & 15);
      #pragma unroll
      for (int r = 0; r < 8; ++r)
        oW[(size_t)(b * T_ + m0 + r) * HID_ + col] = oc[r];
    }

    // ---- state update: S += k^T @ v ----
    #pragma unroll
    for (int ks = 0; ks < 2; ++ks) {
      v16bf bF[2];
      #pragma unroll
      for (int ni = 0; ni < 2; ++ni) bF[ni] = frag_b(vT, 64, ni * 16, ks * 32, lane);
      #pragma unroll
      for (int mi = 0; mi < 2; ++mi) {
        v16bf aF = frag_a(kT, 64, (2 * wave + mi) * 16, ks * 32, lane);
        S[mi][0] = wmma_bf16(aF, bF[0], S[mi][0]);
        S[mi][1] = wmma_bf16(aF, bF[1], S[mi][1]);
      }
    }
    __syncthreads();
  }
}

// ---------------------------------------------------------------------------
// rmsnorm over 1024 f32, emits bf16 for the final GEMM.
__global__ __launch_bounds__(256)
void rmsnorm_k(const float* __restrict__ x, u16t* __restrict__ y) {
  __shared__ float sred[256];
  __shared__ float sinv;
  const int tid = threadIdx.x;
  const size_t base = (size_t)blockIdx.x * HID_;
  float a0 = x[base + tid * 4 + 0];
  float a1 = x[base + tid * 4 + 1];
  float a2 = x[base + tid * 4 + 2];
  float a3 = x[base + tid * 4 + 3];
  sred[tid] = a0 * a0 + a1 * a1 + a2 * a2 + a3 * a3;
  __syncthreads();
  for (int s = 128; s > 0; s >>= 1) {
    if (tid < s) sred[tid] += sred[tid + s];
    __syncthreads();
  }
  if (tid == 0) sinv = rsqrtf(sred[0] * (1.0f / (float)HID_) + 1e-5f);
  __syncthreads();
  float inv = sinv;
  u32t* yp = (u32t*)y;
  yp[(base >> 1) + tid * 2 + 0] = pack2bf(a0 * inv, a1 * inv);
  yp[(base >> 1) + tid * 2 + 1] = pack2bf(a2 * inv, a3 * inv);
}

// f32 -> bf16, processed as pairs (n2 = element count / 2)
__global__ void cvt_f32_bf16(const float* __restrict__ in, u16t* __restrict__ out,
                             int n2) {
  int i = blockIdx.x * blockDim.x + threadIdx.x;
  for (; i < n2; i += gridDim.x * blockDim.x) {
    float2 v = ((const float2*)in)[i];
    ((u32t*)out)[i] = pack2bf(v.x, v.y);
  }
}

// ---------------------------------------------------------------------------
extern "C" void kernel_launch(void* const* d_in, const int* in_sizes, int n_in,
                              void* d_out, int out_size, void* d_ws, size_t ws_size,
                              hipStream_t stream) {
  (void)in_sizes; (void)n_in; (void)out_size; (void)ws_size;
  const float* x    = (const float*)d_in[0];
  const float* Wq   = (const float*)d_in[1];
  const float* Wk   = (const float*)d_in[2];
  const float* Wv   = (const float*)d_in[3];
  const float* Wo   = (const float*)d_in[4];
  const float* fmqw = (const float*)d_in[5];
  const float* fmqb = (const float*)d_in[6];
  const float* fmkw = (const float*)d_in[7];
  const float* fmkb = (const float*)d_in[8];

  char* ws = (char*)d_ws;
  const size_t MB = 1ull << 20;
  u16t*  qb   = (u16t*)(ws + 0 * MB);     // q proj bf16 (32MB)
  u16t*  kb   = (u16t*)(ws + 32 * MB);    // k proj bf16 (32MB)
  u16t*  vb   = (u16t*)(ws + 64 * MB);    // v proj bf16 (32MB)
  u16t*  pq   = (u16t*)(ws + 96 * MB);    // phi_q bf16 (64MB)
  u16t*  pk   = (u16t*)(ws + 160 * MB);   // phi_k bf16 (64MB)
  u16t*  xb   = (u16t*)(ws + 224 * MB);   // x bf16 (32MB)
  u16t*  wqb  = (u16t*)(ws + 256 * MB);   // weights bf16 (2MB each)
  u16t*  wkb  = (u16t*)(ws + 258 * MB);
  u16t*  wvb  = (u16t*)(ws + 260 * MB);
  u16t*  wob  = (u16t*)(ws + 262 * MB);
  u16t*  fqwb = (u16t*)(ws + 264 * MB);   // 32KB
  u16t*  fkwb = (u16t*)(ws + 264 * MB + (64 << 10));
  u16t*  onb  = (u16t*)(ws + 265 * MB);   // normalized o bf16 (32MB)
  float* oW   = (float*)(ws + 0 * MB);    // [B,T,1024] f32 (64MB), reuses qb+kb
  float* out  = (float*)d_out;

  const int M = B_ * T_;                  // 16384
  dim3 blk(256);
  dim3 ggemm(M / 128, HID_ / 128);

  // One-time f32 -> bf16 conversions (activations + weights).
  cvt_f32_bf16<<<dim3(4096), blk, 0, stream>>>(x, xb, (M * HID_) / 2);
  cvt_f32_bf16<<<dim3(512), blk, 0, stream>>>(Wq, wqb, (HID_ * HID_) / 2);
  cvt_f32_bf16<<<dim3(512), blk, 0, stream>>>(Wk, wkb, (HID_ * HID_) / 2);
  cvt_f32_bf16<<<dim3(512), blk, 0, stream>>>(Wv, wvb, (HID_ * HID_) / 2);
  cvt_f32_bf16<<<dim3(512), blk, 0, stream>>>(Wo, wob, (HID_ * HID_) / 2);
  cvt_f32_bf16<<<dim3(32), blk, 0, stream>>>(fmqw, fqwb, (128 * 128) / 2);
  cvt_f32_bf16<<<dim3(32), blk, 0, stream>>>(fmkw, fkwb, (128 * 128) / 2);

  // Projections (bf16 in, bf16 out).
  gemm_wmma<true><<<ggemm, blk, 0, stream>>>(xb, wqb, qb, M, HID_, HID_);
  gemm_wmma<true><<<ggemm, blk, 0, stream>>>(xb, wkb, kb, M, HID_, HID_);
  gemm_wmma<true><<<ggemm, blk, 0, stream>>>(xb, wvb, vb, M, HID_, HID_);

  const size_t hh_smem = 32768 + 16384 + 32768 + 1024;  // 82944 B
  // q path folds in attention scale (2*DQK)^-0.5 = 1/16
  hedgehog_wmma<<<dim3((M * H_) / 64), blk, hh_smem, stream>>>(qb, fqwb, fmqb, pq,
                                                              0.0625f);
  hedgehog_wmma<<<dim3((M * H_) / 64), blk, hh_smem, stream>>>(kb, fkwb, fmkb, pk,
                                                              1.0f);

  const size_t at_smem = (size_t)(64 * 256 + 64 * 256 + 256 * 64 + 32 * 64 +
                                  32 * 256 + 64 * 64) * 2;  // 126976 B
  chunk_attn_wmma<<<dim3(B_ * H_ * 4), blk, at_smem, stream>>>(pq, pk, vb, oW);

  rmsnorm_k<<<dim3(M), blk, 0, stream>>>(oW, onb);
  gemm_wmma<false><<<ggemm, blk, 0, stream>>>(onb, wob, out, M, HID_, HID_);
}
